// MultiHeadAttention_58007828300263
// MI455X (gfx1250) — matrix operand order repaired and checked by reading, not compiled, not run on hardware
//
#include <hip/hip_runtime.h>
#include <hip/hip_bf16.h>
#include <stdint.h>

// ---------------------------------------------------------------------------
// MI455X (gfx1250) multi-head attention, bf16 WMMA pipeline + async-LDS copies
//   B=2, S=2048, HIDDEN=1024, HEADS=16, HEAD_DIM=64
// d_out = [output (2*2048*1024 f32)] ++ [attn (2*16*2048*2048 f32)]
// ---------------------------------------------------------------------------

#define HID    1024
#define SEQ    2048
#define NHEADS 16
#define HD     64
#define STRIPE_BYTES (16 * SEQ * 4)   // 128 KB score stripe in LDS
#define KTILE_BYTES  2048             // one 16x64 bf16 K tile

typedef __attribute__((ext_vector_type(16))) __bf16 bf16x16;
typedef __attribute__((ext_vector_type(8)))  float  f32x8;

union BFrag {
    uint32_t u[8];
    bf16x16  v;
};

__device__ __forceinline__ uint16_t f2bf(float f) {
    uint32_t u = __float_as_uint(f);
    u += 0x7FFFu + ((u >> 16) & 1u);       // round-to-nearest-even
    return (uint16_t)(u >> 16);
}

__device__ __forceinline__ uint32_t pack_bf2(float lo, float hi) {
    return (uint32_t)f2bf(lo) | ((uint32_t)f2bf(hi) << 16);
}

// A-fragment (16x32 bf16, MxK): lane m = lane&15, half = lane>>4.
// VGPR v holds packed {K, K+1} with K = 16*half + 2*v.
__device__ __forceinline__ int a_koff(int v, int half) {
    return (half << 4) + (v << 1);
}

// --- CDNA5 async global->LDS copy (ASYNCcnt-tracked), inline asm ------------
// Copies one 16-row x 128-byte tile (16x64 bf16) with 4 b128 async ops.
// lds_addr: low 32 bits of a flat LDS pointer == LDS byte address (ISA §10.2).
__device__ __forceinline__ void async_ktile_load(uint32_t lds_base,
                                                 const uint16_t* gsrc,
                                                 int lane) {
    const int r = lane >> 3;   // 0..3  (row within 4-row group)
    const int c = lane & 7;    // 0..7  (16B chunk within 128B row)
#pragma unroll
    for (int i = 0; i < 4; ++i) {
        uint32_t laddr = lds_base + (uint32_t)((i * 4 + r) * 128 + c * 16);
        const uint8_t* gaddr = (const uint8_t*)gsrc +
                               (size_t)(i * 4 + r) * (HID * 2) + c * 16;
        asm volatile("global_load_async_to_lds_b128 %0, %1, off"
                     :: "v"(laddr), "v"(gaddr)
                     : "memory");
    }
}
__device__ __forceinline__ void wait_async_le(int keep4) {
    if (keep4) asm volatile("s_wait_asynccnt 0x4" ::: "memory");
    else       asm volatile("s_wait_asynccnt 0x0" ::: "memory");
}

// ---------------------------------------------------------------------------
// Stage 1: f32 -> bf16 conversion
// ---------------------------------------------------------------------------
__global__ void cvt_f32_to_bf16(const float* __restrict__ src,
                                uint16_t* __restrict__ dst, int n) {
    int i = blockIdx.x * blockDim.x + threadIdx.x;
    if (i < n) dst[i] = f2bf(src[i]);
}

// ---------------------------------------------------------------------------
// Stage 2/4: Y = A @ W^T + bias.  A:[M,K] bf16, W:[N,K] bf16.
// One wave computes a 16x64 output tile (4 WMMA accumulators along N).
// mode 0: bf16 row-major out.  mode 1: f32 row-major out.
// mode 2: bf16 out transposed per head: Vt[b][head][d][pos] (for P@V B-frags).
// ---------------------------------------------------------------------------
__global__ void gemm_bf16_wmma(const uint16_t* __restrict__ A,
                               const uint16_t* __restrict__ W,
                               const float* __restrict__ bias,
                               uint16_t* __restrict__ out_bf,
                               float* __restrict__ out_f,
                               int Kdim, int Npitch, int mode) {
    const int lane = threadIdx.x & 31;
    const int mrow = lane & 15;
    const int half = lane >> 4;
    const int m0   = blockIdx.x * 16;
    const int n0   = blockIdx.y * 64;

    f32x8 acc[4];
#pragma unroll
    for (int t = 0; t < 4; ++t)
#pragma unroll
        for (int e = 0; e < 8; ++e) acc[t][e] = 0.0f;

    const uint16_t* arow = A + (size_t)(m0 + mrow) * Kdim;

    for (int kb = 0; kb < Kdim; kb += 32) {
        if (kb + 32 < Kdim) __builtin_prefetch(arow + kb + 32, 0, 1);

        BFrag a;
#pragma unroll
        for (int v = 0; v < 8; ++v)
            a.u[v] = *(const uint32_t*)(arow + kb + a_koff(v, half));

        // load all four B fragments first -> one big load clause, one wait
        BFrag b[4];
#pragma unroll
        for (int t = 0; t < 4; ++t) {
            const uint16_t* wrow =
                W + (size_t)(n0 + t * 16 + mrow) * Kdim + kb + (half << 4);
#pragma unroll
            for (int v = 0; v < 8; ++v)
                b[t].u[v] = *(const uint32_t*)(wrow + 2 * v);
        }
#pragma unroll
        for (int t = 0; t < 4; ++t)
            acc[t] = __builtin_amdgcn_wmma_f32_16x16x32_bf16(
                false, a.v, false, b[t].v, (short)0, acc[t], false, false);
    }

#pragma unroll
    for (int t = 0; t < 4; ++t) {
        const int n = n0 + t * 16 + mrow;
        const float bv = bias[n];
#pragma unroll
        for (int v = 0; v < 8; ++v) {
            const int m = m0 + v + 8 * half;
            const float val = acc[t][v] + bv;
            if (mode == 1) {
                out_f[(size_t)m * Npitch + n] = val;
            } else if (mode == 2) {
                const int bb  = m >> 11;          // m / SEQ
                const int pos = m & (SEQ - 1);
                const int hh  = n >> 6;           // n / HD
                const int dc  = n & (HD - 1);
                out_bf[(((size_t)bb * NHEADS + hh) * HD + dc) * SEQ + pos] =
                    f2bf(val);
            } else {
                out_bf[(size_t)m * Npitch + n] = f2bf(val);
            }
        }
    }
}

// ---------------------------------------------------------------------------
// Stage 3: fused attention for one (batch, head, 16-row q tile).
//   K tiles double-buffered in LDS via global_load_async_to_lds_b128,
//   scores (WMMA) -> 128KB LDS stripe -> softmax in LDS
//   -> attn streamed to HBM with NT stores -> context = P @ Vt (WMMA).
// One wave per workgroup; dynamic LDS = 128KB stripe + 2 x 2KB K tiles.
// ---------------------------------------------------------------------------
__global__ void attn_fused_wmma(const uint16_t* __restrict__ Qb,  // [B*S, HID]
                                const uint16_t* __restrict__ Kb,  // [B*S, HID]
                                const uint16_t* __restrict__ Vt,  // [B,H,HD,S]
                                float* __restrict__ attn,         // [B,H,S,S]
                                uint16_t* __restrict__ ctx) {     // [B*S, HID]
    extern __shared__ char smem[];
    float* sc = (float*)smem;                       // 16 x SEQ score stripe
    const int lane = threadIdx.x & 31;
    const int mrow = lane & 15;
    const int half = lane >> 4;
    const int qt = blockIdx.x;
    const int h  = blockIdx.y;
    const int bb = blockIdx.z;
    const int q0 = qt * 16;
    const size_t rowbase = (size_t)bb * SEQ;

    // --- preload Q A-fragments (16 x 64, two K=32 chunks) -------------------
    BFrag qa[2];
    {
        const uint16_t* qrow = Qb + (rowbase + q0 + mrow) * HID + h * HD;
#pragma unroll
        for (int c = 0; c < 2; ++c)
#pragma unroll
            for (int v = 0; v < 8; ++v)
                qa[c].u[v] =
                    *(const uint32_t*)(qrow + c * 32 + a_koff(v, half));
    }

    const float scale = 0.125f;  // 1/sqrt(64)
    const uint16_t* khead = Kb + rowbase * HID + h * HD;

    // --- scores: S = (Q K^T) * scale into LDS, async-pipelined K tiles ------
    async_ktile_load((uint32_t)(uintptr_t)(smem + STRIPE_BYTES), khead, lane);
    for (int kt = 0; kt < SEQ / 16; ++kt) {
        const int more = (kt + 1 < SEQ / 16);
        if (more)
            async_ktile_load(
                (uint32_t)(uintptr_t)(smem + STRIPE_BYTES +
                                      ((kt + 1) & 1) * KTILE_BYTES),
                khead + (size_t)(kt + 1) * 16 * HID, lane);
        wait_async_le(more);  // tile kt resident; tile kt+1 in flight

        const uint16_t* ktp =
            (const uint16_t*)(smem + STRIPE_BYTES + (kt & 1) * KTILE_BYTES);
        f32x8 c;
#pragma unroll
        for (int e = 0; e < 8; ++e) c[e] = 0.0f;
#pragma unroll
        for (int ch = 0; ch < 2; ++ch) {
            BFrag b;  // B[kd][n] = Ktile[n][kd], rows of 64 elements in LDS
#pragma unroll
            for (int v = 0; v < 8; ++v)
                b.u[v] = *(const uint32_t*)(ktp + mrow * 64 + ch * 32 +
                                            (half << 4) + 2 * v);
            c = __builtin_amdgcn_wmma_f32_16x16x32_bf16(
                false, qa[ch].v, false, b.v, (short)0, c, false, false);
        }
#pragma unroll
        for (int v = 0; v < 8; ++v)
            sc[(v + 8 * half) * SEQ + kt * 16 + mrow] = c[v] * scale;
    }
    __syncthreads();

    // --- softmax per row, stream attn out with non-temporal stores ----------
    float* attn_base = attn + (((size_t)bb * NHEADS + h) * SEQ + q0) * SEQ;
    for (int r = 0; r < 16; ++r) {
        float* row = sc + r * SEQ;
        float mx = -3.402823466e38f;
        for (int col = lane; col < SEQ; col += 32) mx = fmaxf(mx, row[col]);
#pragma unroll
        for (int off = 16; off > 0; off >>= 1)
            mx = fmaxf(mx, __shfl_xor(mx, off, 32));
        float sum = 0.0f;
        for (int col = lane; col < SEQ; col += 32) {
            float e = __expf(row[col] - mx);
            row[col] = e;
            sum += e;
        }
#pragma unroll
        for (int off = 16; off > 0; off >>= 1)
            sum += __shfl_xor(sum, off, 32);
        const float inv = 1.0f / sum;
        for (int col = lane; col < SEQ; col += 32) {
            float p = row[col] * inv;
            row[col] = p;
            // 537 MB streaming output: NT store, don't pollute the 192MB L2
            __builtin_nontemporal_store(p, &attn_base[(size_t)r * SEQ + col]);
        }
    }
    __syncthreads();

    // --- context: C(16x64) = P(16xSEQ) @ V(SEQx64), B-frags from Vt ---------
    f32x8 acc[4];
#pragma unroll
    for (int t = 0; t < 4; ++t)
#pragma unroll
        for (int e = 0; e < 8; ++e) acc[t][e] = 0.0f;

    const uint16_t* vthead = Vt + ((size_t)bb * NHEADS + h) * HD * SEQ;

    for (int kb = 0; kb < SEQ; kb += 32) {
        BFrag a;  // P row mrow, fp32 in LDS -> bf16 pack on the fly
        const float* prow = sc + mrow * SEQ + kb;
#pragma unroll
        for (int v = 0; v < 8; ++v) {
            int ko = a_koff(v, half);
            a.u[v] = pack_bf2(prow[ko], prow[ko + 1]);
        }
        BFrag b[4];
#pragma unroll
        for (int t = 0; t < 4; ++t) {
            // B[kd][n] = V[kd][t*16+mrow] = Vt[t*16+mrow][kd] (contig in kd)
            const uint16_t* vrow =
                vthead + (size_t)(t * 16 + mrow) * SEQ + kb + (half << 4);
#pragma unroll
            for (int v = 0; v < 8; ++v)
                b[t].u[v] = *(const uint32_t*)(vrow + 2 * v);
        }
#pragma unroll
        for (int t = 0; t < 4; ++t)
            acc[t] = __builtin_amdgcn_wmma_f32_16x16x32_bf16(
                false, a.v, false, b[t].v, (short)0, acc[t], false, false);
    }

#pragma unroll
    for (int t = 0; t < 4; ++t)
#pragma unroll
        for (int v = 0; v < 8; ++v) {
            const int m = q0 + v + 8 * half;
            const int n = h * HD + t * 16 + mrow;
            ctx[(rowbase + m) * HID + n] = f2bf(acc[t][v]);
        }
}

// ---------------------------------------------------------------------------
// Host-side launch
// ---------------------------------------------------------------------------
extern "C" void kernel_launch(void* const* d_in, const int* in_sizes, int n_in,
                              void* d_out, int out_size, void* d_ws,
                              size_t ws_size, hipStream_t stream) {
    const float* q  = (const float*)d_in[0];
    const float* Wq = (const float*)d_in[1];
    const float* bq = (const float*)d_in[2];
    const float* Wk = (const float*)d_in[3];
    const float* bk = (const float*)d_in[4];
    const float* Wv = (const float*)d_in[5];
    const float* bv = (const float*)d_in[6];
    const float* Wo = (const float*)d_in[7];
    const float* bo = (const float*)d_in[8];

    const int Bsz = 2;
    const size_t M = (size_t)Bsz * SEQ;  // 4096 rows

    // workspace carve-up (bf16 buffers), 48 MB total
    uint16_t* Xbf = (uint16_t*)d_ws;          // [M, HID]
    uint16_t* Wqb = Xbf + M * HID;            // [HID, HID]
    uint16_t* Wkb = Wqb + (size_t)HID * HID;
    uint16_t* Wvb = Wkb + (size_t)HID * HID;
    uint16_t* Wob = Wvb + (size_t)HID * HID;
    uint16_t* Qbf = Wob + (size_t)HID * HID;  // [M, HID]
    uint16_t* Kbf = Qbf + M * HID;            // [M, HID]
    uint16_t* Vt  = Kbf + M * HID;            // [B, NHEADS, HD, SEQ]
    uint16_t* Ctx = Vt  + M * HID;            // [M, HID]

    float* out  = (float*)d_out;              // [M, HID]
    float* attn = out + M * HID;              // [B, NHEADS, SEQ, SEQ]

    // Stage 1: f32 -> bf16
    {
        int nX = (int)(M * HID);
        cvt_f32_to_bf16<<<(nX + 255) / 256, 256, 0, stream>>>(q, Xbf, nX);
        int nW = HID * HID;
        dim3 gw((nW + 255) / 256);
        cvt_f32_to_bf16<<<gw, 256, 0, stream>>>(Wq, Wqb, nW);
        cvt_f32_to_bf16<<<gw, 256, 0, stream>>>(Wk, Wkb, nW);
        cvt_f32_to_bf16<<<gw, 256, 0, stream>>>(Wv, Wvb, nW);
        cvt_f32_to_bf16<<<gw, 256, 0, stream>>>(Wo, Wob, nW);
    }

    // Stage 2: Q/K/V projections (bf16 WMMA GEMM); V written head-transposed
    {
        dim3 gg((unsigned)(M / 16), HID / 64);
        gemm_bf16_wmma<<<gg, 32, 0, stream>>>(Xbf, Wqb, bq, Qbf, nullptr,
                                              HID, HID, 0);
        gemm_bf16_wmma<<<gg, 32, 0, stream>>>(Xbf, Wkb, bk, Kbf, nullptr,
                                              HID, HID, 0);
        gemm_bf16_wmma<<<gg, 32, 0, stream>>>(Xbf, Wvb, bv, Vt, nullptr,
                                              HID, HID, 2);
    }

    // Stage 3: fused attention (async K tiles + scores + softmax + context)
    {
        dim3 ga(SEQ / 16, NHEADS, Bsz);
        size_t lds = STRIPE_BYTES + 2 * KTILE_BYTES;  // 132 KB
        attn_fused_wmma<<<ga, 32, lds, stream>>>(Qbf, Kbf, Vt, attn, Ctx);
    }

    // Stage 4: output projection (f32 out + bias)
    {
        dim3 gg((unsigned)(M / 16), HID / 64);
        gemm_bf16_wmma<<<gg, 32, 0, stream>>>(Ctx, Wob, bo, nullptr, out,
                                              HID, HID, 1);
    }
}
